// RotationAttention_10307921510517
// MI455X (gfx1250) — matrix operand order repaired and checked by reading, not compiled, not run on hardware
//
#include <hip/hip_runtime.h>
#include <hip/hip_bf16.h>

// ---------------------------------------------------------------------------
// Problem constants (match reference)
// ---------------------------------------------------------------------------
#define BB   4
#define TT   2048
#define CC   2048
#define HH   16
#define DKK  128
#define RR   64
#define OFF  512
#define SS   (OFF + TT)          // 2560
#define MROWS (BB * TT)          // 8192
#define FCAT (2 * HH * DKK + CC / 2)  // 5120

typedef __attribute__((ext_vector_type(16))) __bf16 v16bf;
typedef __attribute__((ext_vector_type(8)))  float  v8f;

union AFrag { v16bf v; uint4 q[2]; };

__device__ __forceinline__ v8f wmma_bf16(v16bf a, v16bf b, v8f c) {
  return __builtin_amdgcn_wmma_f32_16x16x32_bf16(false, a, false, b, (short)0, c,
                                                 false, false);
}

__device__ __forceinline__ unsigned short f2bf(float f) {
  unsigned int u = __float_as_uint(f);
  unsigned int r = (u + 0x7fffu + ((u >> 16) & 1u)) >> 16;
  return (unsigned short)r;
}

// ---------------------------------------------------------------------------
// CDNA5 async global->LDS copy (ASYNCcnt path). LDS operand is the wave-
// relative LDS byte address == low 32 bits of the generic shared pointer
// ---------------------------------------------------------------------------
__device__ __forceinline__ void async_g2l_b128(const void* gptr, void* lptr) {
  unsigned lds_off = (unsigned)(size_t)lptr;
  unsigned long long ga = (unsigned long long)(size_t)gptr;
  asm volatile("global_load_async_to_lds_b128 %0, %1, off"
               :
               : "v"(lds_off), "v"(ga)
               : "memory");
}

__device__ __forceinline__ void wait_asynccnt0() {
  asm volatile("s_wait_asynccnt 0x0" ::: "memory");
}

// ---------------------------------------------------------------------------
// Elementwise f32 -> bf16 convert
// ---------------------------------------------------------------------------
__global__ void cvt_bf16_kernel(const float* __restrict__ src,
                                unsigned short* __restrict__ dst, int n) {
  int i = blockIdx.x * 256 + threadIdx.x;
  if (i < n) dst[i] = f2bf(src[i]);
}

// ---------------------------------------------------------------------------
// KV-cache scatter: cached_k [B,H,OFF,DK] -> outK f32 [B,H,S,DK] + kb bf16
// ---------------------------------------------------------------------------
__global__ void cache_k_kernel(const float* __restrict__ ck,
                               float* __restrict__ outK,
                               unsigned short* __restrict__ kb) {
  size_t i = (size_t)blockIdx.x * 256 + threadIdx.x;  // B*H*OFF*DK elements
  int d = (int)(i & (DKK - 1));
  size_t r = i >> 7;
  int s = (int)(r & (OFF - 1));
  size_t bh = r >> 9;
  size_t o = (bh * SS + s) * DKK + d;
  float v = ck[i];
  outK[o] = v;
  kb[o] = f2bf(v);
}

// cached_theta [B,H,OFF,R] -> outTh f32 [B,H,S,R] + thetaT bf16 [B*H,R,S]
__global__ void cache_t_kernel(const float* __restrict__ ct,
                               float* __restrict__ outT,
                               unsigned short* __restrict__ thT) {
  size_t i = (size_t)blockIdx.x * 256 + threadIdx.x;  // B*H*OFF*R elements
  int r = (int)(i & (RR - 1));
  size_t q = i >> 6;
  int s = (int)(q & (OFF - 1));
  size_t bh = q >> 9;
  float v = ct[i];
  outT[(bh * SS + s) * RR + r] = v;
  thT[(bh * RR + r) * SS + s] = f2bf(v);
}

// ---------------------------------------------------------------------------
// Tiled bf16 GEMM:  C[M,N] = A[M,K] @ Bt[N,K]^T   (f32 accumulate via WMMA)
// 256 threads = 8 waves; block tile 128x128; wave tile 32x64; K-step 32.
// Tiles staged via GLOBAL_LOAD_ASYNC_TO_LDS (ASYNCcnt), consumed via
// ds_load_b128 fragments feeding v_wmma_f32_16x16x32_bf16.
// mode 0: plain f32 store to outY[M,N]
// mode 1: QK/theta scatter epilogue (projection GEMM)
// ---------------------------------------------------------------------------
__global__ void gemm_bf16_kernel(const unsigned short* __restrict__ A,
                                 const unsigned short* __restrict__ Bt,
                                 int K, int N, int mode,
                                 float* __restrict__ outY,
                                 float* __restrict__ outK,
                                 float* __restrict__ outTh,
                                 unsigned short* __restrict__ qb,
                                 unsigned short* __restrict__ kb,
                                 unsigned short* __restrict__ thT) {
  __shared__ unsigned short As[128][40];  // 128 x 32 (+8 pad)
  __shared__ unsigned short Bs[128][40];

  const int tid  = threadIdx.x;
  const int wid  = tid >> 5;
  const int lane = tid & 31;
  const int ln   = lane & 15;
  const int lh   = lane >> 4;
  const int m0 = blockIdx.x * 128;
  const int n0 = blockIdx.y * 128;
  const int wm = (wid >> 1) * 32;   // wave M offset within block
  const int wn = (wid & 1) * 64;    // wave N offset within block

  v8f acc[2][4];
#pragma unroll
  for (int i = 0; i < 2; ++i)
#pragma unroll
    for (int j = 0; j < 4; ++j) acc[i][j] = (v8f)0.0f;

  const int row = tid >> 1;
  const int ch  = (tid & 1) * 16;

  for (int k0 = 0; k0 < K; k0 += 32) {
    const unsigned short* sa = A  + (size_t)(m0 + row) * K + k0 + ch;
    const unsigned short* sb = Bt + (size_t)(n0 + row) * K + k0 + ch;
    __builtin_prefetch(sa + 32, 0, 1);
    __builtin_prefetch(sb + 32, 0, 1);
    async_g2l_b128(sa,     &As[row][ch]);
    async_g2l_b128(sa + 8, &As[row][ch + 8]);
    async_g2l_b128(sb,     &Bs[row][ch]);
    async_g2l_b128(sb + 8, &Bs[row][ch + 8]);
    wait_asynccnt0();
    __syncthreads();

    AFrag af[2];
#pragma unroll
    for (int i = 0; i < 2; ++i) {
      const unsigned short* p = &As[wm + i * 16 + ln][8 * lh];
      af[i].q[0] = *(const uint4*)p;
      af[i].q[1] = *(const uint4*)(p + 16);
    }
#pragma unroll
    for (int j = 0; j < 4; ++j) {
      AFrag bf;
      const unsigned short* p = &Bs[wn + j * 16 + ln][8 * lh];
      bf.q[0] = *(const uint4*)p;
      bf.q[1] = *(const uint4*)(p + 16);
#pragma unroll
      for (int i = 0; i < 2; ++i)
        acc[i][j] = wmma_bf16(af[i].v, bf.v, acc[i][j]);
    }
    __syncthreads();
  }

  // Epilogue: element (m = e + 8*lh, n = ln) of each 16x16 tile
#pragma unroll
  for (int i = 0; i < 2; ++i) {
#pragma unroll
    for (int j = 0; j < 4; ++j) {
      const int rbase = m0 + wm + i * 16;
      const int cbase = n0 + wn + j * 16;
#pragma unroll
      for (int e = 0; e < 8; ++e) {
        const int m = e + 8 * lh;
        const int g = rbase + m;       // global row in [0, M)
        const int f = cbase + ln;      // global col (output feature)
        const float v = acc[i][j][e];
        if (mode == 0) {
          outY[(size_t)g * N + f] = v;
        } else {
          const int b = g >> 11;       // g / T
          const int t = g & (TT - 1);
          if (f < 2048) {              // Q -> bf16 head-major scratch
            const int h = f >> 7, d = f & 127;
            qb[(((size_t)(b * HH + h) * TT + t) * DKK) + d] = f2bf(v);
          } else if (f < 4096) {       // K_new -> f32 output + bf16 scratch
            const int f2 = f - 2048;
            const int h = f2 >> 7, d = f2 & 127;
            const size_t idx = ((size_t)(b * HH + h) * SS + OFF + t) * DKK + d;
            outK[idx] = v;
            kb[idx] = f2bf(v);
          } else {                     // theta_new -> f32 output + bf16 thetaT
            const int f3 = f - 4096;
            const int h = f3 >> 6, r = f3 & 63;
            outTh[((size_t)(b * HH + h) * SS + OFF + t) * RR + r] = v;
            thT[((size_t)(b * HH + h) * RR + r) * SS + OFF + t] = f2bf(v);
          }
        }
      }
    }
  }
}

// ---------------------------------------------------------------------------
// Flash-style causal attention with WMMA:
//   scores = q@k^T / sqrt(128), causal (j <= OFF+i), online softmax,
//   phi = attn @ theta  -> phi f32 [B,T, H*R]
// Block = 128 threads (4 waves); block owns 64 queries of one (b,h);
// waves share 32-key K/thetaT tiles (async-loaded into LDS); each wave
// owns 16 queries.
// ---------------------------------------------------------------------------
__global__ void attn_kernel(const unsigned short* __restrict__ qb,
                            const unsigned short* __restrict__ kb,
                            const unsigned short* __restrict__ thT,
                            float* __restrict__ phi) {
  __shared__ unsigned short kt[32][136];     // 32 keys x 128 d (pad 8)
  __shared__ unsigned short tt[64][40];      // 64 r x 32 keys (pad 8)
  __shared__ unsigned short pt[4][16][40];   // per-wave prob tile scratch

  const int bh   = blockIdx.y;               // b*16 + h
  const int qblk = blockIdx.x * 64;
  const int tid  = threadIdx.x;
  const int wid  = tid >> 5;
  const int lane = tid & 31;
  const int ln   = lane & 15;
  const int lh   = lane >> 4;

  // Load this wave's 16-query A-fragments (K = 128 -> 4 chunks of 32)
  const unsigned short* qrow =
      qb + ((size_t)bh * TT + qblk + wid * 16) * DKK;
  AFrag qa[4];
#pragma unroll
  for (int c = 0; c < 4; ++c) {
    const unsigned short* p = qrow + (size_t)ln * DKK + c * 32 + 8 * lh;
    qa[c].q[0] = *(const uint4*)p;
    qa[c].q[1] = *(const uint4*)(p + 16);
  }

  v8f phacc[4];
#pragma unroll
  for (int rt = 0; rt < 4; ++rt) phacc[rt] = (v8f)0.0f;
  float mrun[8], lrun[8];
#pragma unroll
  for (int i = 0; i < 8; ++i) { mrun[i] = -3.0e38f; lrun[i] = 0.0f; }

  const unsigned short* kbase = kb  + (size_t)bh * SS * DKK;
  const unsigned short* tbase = thT + (size_t)bh * RR * SS;
  const float scl = 0.08838834764831845f;    // 1/sqrt(128)
  const int sEnd = OFF + qblk + 64;          // multiple of 32, <= S

  for (int s = 0; s < sEnd; s += 32) {
    // --- cooperative async tile loads (global -> LDS, ASYNCcnt) ---
#pragma unroll
    for (int i = 0; i < 4; ++i) {            // k tile: 512 chunks of 16B
      const int chunk = tid + i * 128;
      const int r = chunk >> 4;              // 16 chunks per 128-elem row
      const int c = (chunk & 15) * 8;
      async_g2l_b128(kbase + (size_t)(s + r) * DKK + c, &kt[r][c]);
    }
#pragma unroll
    for (int i = 0; i < 2; ++i) {            // thetaT tile: 256 chunks of 16B
      const int chunk = tid + i * 128;
      const int r = chunk >> 2;              // 4 chunks per 32-elem row
      const int c = (chunk & 3) * 8;
      async_g2l_b128(tbase + (size_t)r * SS + s + c, &tt[r][c]);
    }
    wait_asynccnt0();
    __syncthreads();

    // --- scores: 2 key-subtiles x 4 K-chunks ---
    v8f sc[2];
    sc[0] = (v8f)0.0f; sc[1] = (v8f)0.0f;
#pragma unroll
    for (int c = 0; c < 4; ++c) {
#pragma unroll
      for (int nt = 0; nt < 2; ++nt) {
        AFrag bf;
        const unsigned short* p = &kt[nt * 16 + ln][c * 32 + 8 * lh];
        bf.q[0] = *(const uint4*)p;
        bf.q[1] = *(const uint4*)(p + 16);
        sc[nt] = wmma_bf16(qa[c].v, bf.v, sc[nt]);
      }
    }

    // --- mask + online softmax (f32) ---
#pragma unroll
    for (int i = 0; i < 8; ++i) {
      const int m  = i + 8 * lh;
      const int qg = qblk + wid * 16 + m;
      const int lim = OFF + qg;
      const float v0 = (s + ln      <= lim) ? sc[0][i] * scl : -3.0e38f;
      const float v1 = (s + 16 + ln <= lim) ? sc[1][i] * scl : -3.0e38f;
      float rmax = fmaxf(v0, v1);
#pragma unroll
      for (int msk = 1; msk < 16; msk <<= 1)
        rmax = fmaxf(rmax, __shfl_xor(rmax, msk, 32));
      const float mnew = fmaxf(mrun[i], rmax);
      const float corr = __expf(mrun[i] - mnew);
      mrun[i] = mnew;
      const float p0 = __expf(v0 - mnew);
      const float p1 = __expf(v1 - mnew);
      float rsum = p0 + p1;
#pragma unroll
      for (int msk = 1; msk < 16; msk <<= 1)
        rsum += __shfl_xor(rsum, msk, 32);
      lrun[i] = lrun[i] * corr + rsum;
#pragma unroll
      for (int rt = 0; rt < 4; ++rt) phacc[rt][i] *= corr;
      pt[wid][m][ln]      = f2bf(p0);
      pt[wid][m][16 + ln] = f2bf(p1);
    }

    // --- phi += p @ theta  (per-wave p tile, DS ops in-order per wave) ---
    AFrag pa;
    {
      const unsigned short* p = &pt[wid][ln][8 * lh];
      pa.q[0] = *(const uint4*)p;
      pa.q[1] = *(const uint4*)(p + 16);
    }
#pragma unroll
    for (int rt = 0; rt < 4; ++rt) {
      AFrag bf;
      const unsigned short* p = &tt[rt * 16 + ln][8 * lh];
      bf.q[0] = *(const uint4*)p;
      bf.q[1] = *(const uint4*)(p + 16);
      phacc[rt] = wmma_bf16(pa.v, bf.v, phacc[rt]);
    }
    __syncthreads();
  }

  // --- normalize and write phi as [B,T, H*R] ---
  const int b = bh >> 4;
  const int h = bh & 15;
#pragma unroll
  for (int rt = 0; rt < 4; ++rt) {
#pragma unroll
    for (int i = 0; i < 8; ++i) {
      const int m = i + 8 * lh;
      const int t = qblk + wid * 16 + m;
      const float v = phacc[rt][i] / lrun[i];
      phi[((size_t)(b * TT + t)) * (CC / 2) + h * RR + rt * 16 + ln] = v;
    }
  }
}

// ---------------------------------------------------------------------------
// Rotation: y pairs = R(phi) * x pairs, emitted as bf16 rows for final GEMM
// ---------------------------------------------------------------------------
__global__ void rot_kernel(const float* __restrict__ x,
                           const float* __restrict__ phi,
                           unsigned short* __restrict__ yb) {
  size_t i = (size_t)blockIdx.x * 256 + threadIdx.x;  // B*T*C/2 pairs
  float ph = phi[i];
  float sn, cs;
  __sincosf(ph, &sn, &cs);
  const float x0 = x[2 * i];
  const float x1 = x[2 * i + 1];
  yb[2 * i]     = f2bf(cs * x0 - sn * x1);
  yb[2 * i + 1] = f2bf(sn * x0 + cs * x1);
}

// ---------------------------------------------------------------------------
// Host launcher
// ---------------------------------------------------------------------------
extern "C" void kernel_launch(void* const* d_in, const int* in_sizes, int n_in,
                              void* d_out, int out_size, void* d_ws,
                              size_t ws_size, hipStream_t stream) {
  const float* x    = (const float*)d_in[0];
  const float* ck   = (const float*)d_in[1];
  const float* ct   = (const float*)d_in[2];
  const float* wqk  = (const float*)d_in[3];
  const float* wv   = (const float*)d_in[4];
  const float* wout = (const float*)d_in[5];

  float* outY  = (float*)d_out;                       // [B,T,C]
  float* outK  = outY + (size_t)BB * TT * CC;         // [B,H,S,DK]
  float* outTh = outK + (size_t)BB * HH * SS * DKK;   // [B,H,S,R]

  char* ws = (char*)d_ws;
  const size_t XB_OFF   = 0;                               // bf16 x   [8192,2048]
  const size_t WCAT_OFF = XB_OFF   + (size_t)MROWS * CC * 2;   // bf16 [5120,2048]
  const size_t WOUT_OFF = WCAT_OFF + (size_t)FCAT * CC * 2;    // bf16 [2048,2048]
  const size_t QB_OFF   = WOUT_OFF + (size_t)CC * CC * 2;      // bf16 q  [B*H,T,DK]
  const size_t KB_OFF   = QB_OFF   + (size_t)BB * HH * TT * DKK * 2;  // bf16 k [B*H,S,DK]
  const size_t THT_OFF  = KB_OFF   + (size_t)BB * HH * SS * DKK * 2;  // bf16 thetaT [B*H,R,S]
  const size_t PHI_OFF  = THT_OFF  + (size_t)BB * HH * RR * SS * 2;   // f32 phi [B,T,C/2]
  const size_t YB_OFF   = PHI_OFF  + (size_t)BB * TT * (CC / 2) * 4;  // bf16 y [8192,2048]

  unsigned short* xb   = (unsigned short*)(ws + XB_OFF);
  unsigned short* wcat = (unsigned short*)(ws + WCAT_OFF);
  unsigned short* wob  = (unsigned short*)(ws + WOUT_OFF);
  unsigned short* qbuf = (unsigned short*)(ws + QB_OFF);
  unsigned short* kbuf = (unsigned short*)(ws + KB_OFF);
  unsigned short* tht  = (unsigned short*)(ws + THT_OFF);
  float*          phi  = (float*)(ws + PHI_OFF);
  unsigned short* yb   = (unsigned short*)(ws + YB_OFF);

  // 1) convert activations + weights to bf16
  {
    int n = MROWS * CC;  // 16,777,216
    cvt_bf16_kernel<<<n / 256, 256, 0, stream>>>(x, xb, n);
  }
  {
    int n = 2 * HH * DKK * CC;  // W_QK: 8,388,608
    cvt_bf16_kernel<<<n / 256, 256, 0, stream>>>(wqk, wcat, n);
  }
  {
    int n = (CC / 2) * CC;      // W_V: 2,097,152 -> rows 4096..5119
    cvt_bf16_kernel<<<n / 256, 256, 0, stream>>>(
        wv, wcat + (size_t)2 * HH * DKK * CC, n);
  }
  {
    int n = CC * CC;            // W_out: 4,194,304
    cvt_bf16_kernel<<<n / 256, 256, 0, stream>>>(wout, wob, n);
  }

  // 2) scatter KV caches into outputs + bf16 scratch
  {
    int n = BB * HH * OFF * DKK;  // 4,194,304
    cache_k_kernel<<<n / 256, 256, 0, stream>>>(ck, outK, kbuf);
  }
  {
    int n = BB * HH * OFF * RR;   // 2,097,152
    cache_t_kernel<<<n / 256, 256, 0, stream>>>(ct, outTh, tht);
  }

  // 3) projection GEMM with scatter epilogue: [8192,2048] x [5120,2048]^T
  {
    dim3 grid(MROWS / 128, FCAT / 128);  // (64, 40)
    gemm_bf16_kernel<<<grid, 256, 0, stream>>>(
        xb, wcat, CC, FCAT, /*mode=*/1, nullptr, outK, outTh, qbuf, kbuf, tht);
  }

  // 4) causal attention -> phi
  {
    dim3 grid(TT / 64, BB * HH);  // (32, 64)
    attn_kernel<<<grid, 128, 0, stream>>>(qbuf, kbuf, tht, phi);
  }

  // 5) rotation
  {
    size_t n = (size_t)BB * TT * (CC / 2);  // 8,388,608 pairs
    rot_kernel<<<(unsigned)(n / 256), 256, 0, stream>>>(x, phi, yb);
  }

  // 6) output projection GEMM: [8192,2048] x [2048,2048]^T -> y
  {
    dim3 grid(MROWS / 128, CC / 128);  // (64, 16)
    gemm_bf16_kernel<<<grid, 256, 0, stream>>>(
        yb, wob, CC, CC, /*mode=*/0, outY, nullptr, nullptr, nullptr, nullptr,
        nullptr);
  }
}
